// LinearAttention_695784702727
// MI455X (gfx1250) — matrix operand order repaired and checked by reading, not compiled, not run on hardware
//
#include <hip/hip_runtime.h>

typedef __attribute__((ext_vector_type(16))) _Float16 v16h;
typedef __attribute__((ext_vector_type(8)))  _Float16 v8h;
typedef __attribute__((ext_vector_type(4)))  _Float16 v4h;
typedef __attribute__((ext_vector_type(2)))  _Float16 v2h;
typedef __attribute__((ext_vector_type(8)))  float    v8f;
typedef _Float16 half_t;

typedef unsigned int u32x4 __attribute__((ext_vector_type(4)));
typedef int          i32x8 __attribute__((ext_vector_type(8)));
typedef int          i32x4 __attribute__((ext_vector_type(4)));

#define BATCH   16
#define DIMC    256
#define NPIX    4096
#define OC3     384
#define HID     128
#define NHEADS  4
#define DH      32
#define GN_EPS  1e-5f

#if __has_builtin(__builtin_amdgcn_tensor_load_to_lds)
#define HAVE_TDM 1
#else
#define HAVE_TDM 0
#endif

// ---- WMMA fragment index helpers (CDNA5 ISA 7.12.2, wave32, f16 16x16x32) ----
// A lane (lo,hi): row=lo, k = {8*hi + 0..7, 16 + 8*hi + 0..7}  (two contiguous runs)
// B lane (lo,hi): col=lo, k = {8*hi + 0..7, 16 + 8*hi + 0..7}  (same as A)
__device__ __forceinline__ int ka_idx(int hi, int j) {
    return (j < 8) ? (8 * hi + j) : (16 + 8 * hi + (j - 8));
}
__device__ __forceinline__ int kb_idx(int hi, int j) {
    return (j < 8) ? (8 * hi + j) : (16 + 8 * hi + (j - 8));
}

__device__ __forceinline__ v16h frag_a_lds(const half_t* la, int row, int hi) {
    v8h a0 = *(const v8h*)(la + row * 32 + 8 * hi);
    v8h a1 = *(const v8h*)(la + row * 32 + 16 + 8 * hi);
    return __builtin_shufflevector(a0, a1, 0, 1, 2, 3, 4, 5, 6, 7,
                                   8, 9, 10, 11, 12, 13, 14, 15);
}
__device__ __forceinline__ v16h frag_b_lds(const half_t* lb, int col, int hi) {
    v8h b0 = *(const v8h*)(lb + col * 32 + 8 * hi);
    v8h b1 = *(const v8h*)(lb + col * 32 + 16 + 8 * hi);
    return __builtin_shufflevector(b0, b1, 0, 1, 2, 3, 4, 5, 6, 7,
                                   8, 9, 10, 11, 12, 13, 14, 15);
}

#if HAVE_TDM
// 2D tile load via Tensor Data Mover (D# per CDNA5 ISA).
// data_size_log2: 0=1B,1=2B,2=4B. tile_d0 = contiguous dim, stride0 in elements.
// 6-arg builtin form (clang-23 / therock-10.0 headers).
__device__ __forceinline__ void tdm_load_2d(unsigned lds_addr, const void* gptr,
                                            unsigned tile_d0, unsigned tile_d1,
                                            unsigned tensor_d0, unsigned tensor_d1,
                                            unsigned stride0, unsigned data_size_log2)
{
    const unsigned long long ga = (unsigned long long)(uintptr_t)gptr;
    u32x4 g0;
    g0[0] = 1u;                                              // count=1, user mode
    g0[1] = lds_addr;                                        // LDS byte address
    g0[2] = (unsigned)(ga & 0xFFFFFFFFu);                    // global_addr[31:0]
    g0[3] = (unsigned)((ga >> 32) & 0x1FFFFFFu) | (2u << 30);// global_addr[56:32], type=2
    i32x8 g1;
    g1[0] = (int)(data_size_log2 << 16);                     // wg_mask=0, data_size
    g1[1] = (int)((tensor_d0 & 0xFFFFu) << 16);              // tensor_dim0[15:0]
    g1[2] = (int)(((tensor_d0 >> 16) & 0xFFFFu) | ((tensor_d1 & 0xFFFFu) << 16));
    g1[3] = (int)(((tensor_d1 >> 16) & 0xFFFFu) | ((tile_d0 & 0xFFFFu) << 16));
    g1[4] = (int)(tile_d1 & 0xFFFFu);                        // tile_dim1, tile_dim2=0
    g1[5] = (int)stride0;                                    // tensor_dim0_stride[31:0]
    g1[6] = 0;
    g1[7] = 0;
    i32x4 z4 = {0, 0, 0, 0};
    i32x8 z8 = {0, 0, 0, 0, 0, 0, 0, 0};
    __builtin_amdgcn_tensor_load_to_lds(g0, g1, z4, z4, z8, 0);
}
#endif

// ============ Kernel 1: qkv = W_qkv[384,256] @ x_b[256,4096], store f16 ============
// Block: 8 waves -> 128 rows x 64 cols. Double-buffered TDM x-tile [32k x 64n] f32 -> LDS
// (DMA of step i+1 overlaps transpose+WMMA of step i); transpose+cvt to [n][k] f16 with
// packed b32 stores; W_qkv tile: float4 loads -> packed v4h LDS stores; ds_load_b128 frags.
__global__ __launch_bounds__(256)
void k_qkv_gemm(const float* __restrict__ x, const float* __restrict__ Wqkv,
                half_t* __restrict__ qkv_h)
{
    __shared__ float  xraw[2][32 * 64]; // 16 KB [k][n], double-buffered TDM dest
    __shared__ half_t lb[64 * 32];      //  4 KB [n][k]
    __shared__ half_t la[128 * 32];     //  8 KB [row][k]

    const int cg = blockIdx.x & 63;                 // 64-col group
    const int rg = (blockIdx.x >> 6) % 3;           // 128-row group of 384
    const int b  = blockIdx.x / 192;
    const int tid  = threadIdx.x;
    const int wid  = tid >> 5;
    const int lane = tid & 31;
    const int lo = lane & 15, hi = lane >> 4;
    const int c0 = cg * 64;
    const int rbase = rg * 128;
    const float* xb = x + (size_t)b * DIMC * NPIX;

#if HAVE_TDM
    if (tid < 32)
        tdm_load_2d((unsigned)(uintptr_t)xraw[0], xb + c0,
                    64u, 32u, (unsigned)NPIX, (unsigned)NPIX, (unsigned)NPIX, 2u);
#endif

    v8f acc[4] = {};
    for (int ks = 0; ks < 8; ++ks) {
        const int k0 = ks * 32;
        const int cur = ks & 1;
#if HAVE_TDM
        if (tid < 32) {
            if (ks < 7) {
                tdm_load_2d((unsigned)(uintptr_t)xraw[cur ^ 1],
                            xb + (size_t)(k0 + 32) * NPIX + c0,
                            64u, 32u, (unsigned)NPIX, (unsigned)NPIX, (unsigned)NPIX, 2u);
                __builtin_amdgcn_s_wait_tensorcnt(1);   // tile ks retired (in-order)
            } else {
                __builtin_amdgcn_s_wait_tensorcnt(0);
            }
        }
#else
        #pragma unroll
        for (int i = 0; i < 8; ++i) {
            const int idx = i * 256 + tid;
            xraw[cur][idx] = xb[(size_t)(k0 + (idx >> 6)) * NPIX + c0 + (idx & 63)];
        }
#endif
        // W_qkv tile -> LDS f16: float4 loads, packed v4h stores (128 rows x 32 k)
        #pragma unroll
        for (int i = 0; i < 4; ++i) {
            const int idx4 = i * 256 + tid;         // 0..1023 float4s
            const int row = idx4 >> 3, koff = (idx4 & 7) * 4;
            const float4 w = *(const float4*)(Wqkv + (size_t)(rbase + row) * DIMC + k0 + koff);
            v4h p = {(half_t)w.x, (half_t)w.y, (half_t)w.z, (half_t)w.w};
            *(v4h*)(la + row * 32 + koff) = p;
        }
        __syncthreads();
        // transpose+cvt x: [k][n] f32 -> [n][k] f16, packed b32 stores
        const float* xr = xraw[cur];
        #pragma unroll
        for (int i = 0; i < 4; ++i) {
            const int idx = i * 256 + tid;          // 0..1023 pairs
            const int n = idx & 63, k = (idx >> 6) * 2;
            v2h p = {(half_t)xr[k * 64 + n], (half_t)xr[(k + 1) * 64 + n]};
            *(v2h*)(lb + n * 32 + k) = p;
        }
        __syncthreads();
        const v16h a = frag_a_lds(la, wid * 16 + lo, hi);
        #pragma unroll
        for (int t = 0; t < 4; ++t) {
            const v16h bf = frag_b_lds(lb, t * 16 + lo, hi);
            acc[t] = __builtin_amdgcn_wmma_f32_16x16x32_f16(false, a, false, bf,
                                                            (short)0, acc[t], false, false);
        }
        __syncthreads();
    }
    half_t* outb = qkv_h + (size_t)b * OC3 * NPIX;
    #pragma unroll
    for (int t = 0; t < 4; ++t) {
        const int col = c0 + t * 16 + lo;
        #pragma unroll
        for (int r = 0; r < 8; ++r) {
            const int orow = rbase + wid * 16 + r + 8 * hi;
            outb[(size_t)orow * NPIX + col] = (half_t)acc[t][r];
        }
    }
}

// ============ Kernel 2: q = softmax over d (32) * scale, in place ============
__global__ __launch_bounds__(256)
void k_q_softmax(half_t* __restrict__ qkv_h)
{
    const int t = blockIdx.x * 256 + threadIdx.x;
    const int n = t & (NPIX - 1);
    const int head = (t >> 12) & 3;
    const int b = t >> 14;
    half_t* base = qkv_h + ((size_t)(b * OC3 + head * DH) * NPIX) + n;
    float v[DH];
    float m = -1e30f;
    #pragma unroll
    for (int d = 0; d < DH; ++d) { v[d] = (float)base[(size_t)d * NPIX]; m = fmaxf(m, v[d]); }
    float s = 0.f;
    #pragma unroll
    for (int d = 0; d < DH; ++d) { v[d] = __expf(v[d] - m); s += v[d]; }
    const float inv = 0.17677669529663689f / s;
    #pragma unroll
    for (int d = 0; d < DH; ++d) base[(size_t)d * NPIX] = (half_t)(v[d] * inv);
}

// ============ Kernel 3: k = softmax over n (4096), one block per row ============
__global__ __launch_bounds__(256)
void k_k_softmax(half_t* __restrict__ qkv_h)
{
    __shared__ float red[256];
    const int row = blockIdx.x;
    const int b = row >> 7;
    const int c = row & 127;
    half_t* base = qkv_h + ((size_t)(b * OC3 + HID + c) * NPIX);
    const int tid = threadIdx.x;
    float v[16];
    float m = -1e30f;
    #pragma unroll
    for (int i = 0; i < 16; ++i) { v[i] = (float)base[tid + i * 256]; m = fmaxf(m, v[i]); }
    red[tid] = m; __syncthreads();
    for (int s = 128; s > 0; s >>= 1) {
        if (tid < s) red[tid] = fmaxf(red[tid], red[tid + s]);
        __syncthreads();
    }
    m = red[0]; __syncthreads();
    float sum = 0.f;
    #pragma unroll
    for (int i = 0; i < 16; ++i) { v[i] = __expf(v[i] - m); sum += v[i]; }
    red[tid] = sum; __syncthreads();
    for (int s = 128; s > 0; s >>= 1) {
        if (tid < s) red[tid] += red[tid + s];
        __syncthreads();
    }
    const float inv = 1.f / red[0];
    #pragma unroll
    for (int i = 0; i < 16; ++i) base[tid + i * 256] = (half_t)(v[i] * inv);
}

// ============ Kernel 4: context[b,h,d,e] = sum_n ksm[d,n]*v[e,n], split-K + atomics ====
__global__ __launch_bounds__(256)
void k_context(const half_t* __restrict__ qkv_h, float* __restrict__ ctx)
{
    const int wave = blockIdx.x * 8 + (threadIdx.x >> 5);
    const int lane = threadIdx.x & 31;
    const int lo = lane & 15, hi = lane >> 4;
    const int chunk = wave & 15;
    const int head  = (wave >> 4) & 3;
    const int b     = wave >> 6;
    const size_t kbase = (size_t)(b * OC3 + HID + head * DH) * NPIX;
    const size_t vbase = (size_t)(b * OC3 + 2 * HID + head * DH) * NPIX;
    const int n0 = chunk * 256;
    v8f acc[2][2] = {};
    for (int ks = 0; ks < 8; ++ks) {
        const int nb = n0 + ks * 32;
        v16h a[2], bb[2];
        #pragma unroll
        for (int rt = 0; rt < 2; ++rt)
            #pragma unroll
            for (int j = 0; j < 16; ++j)
                a[rt][j] = qkv_h[kbase + (size_t)(rt * 16 + lo) * NPIX + nb + ka_idx(hi, j)];
        #pragma unroll
        for (int ct = 0; ct < 2; ++ct)
            #pragma unroll
            for (int j = 0; j < 16; ++j)
                bb[ct][j] = qkv_h[vbase + (size_t)(ct * 16 + lo) * NPIX + nb + kb_idx(hi, j)];
        #pragma unroll
        for (int rt = 0; rt < 2; ++rt)
            #pragma unroll
            for (int ct = 0; ct < 2; ++ct)
                acc[rt][ct] = __builtin_amdgcn_wmma_f32_16x16x32_f16(false, a[rt], false, bb[ct],
                                                                     (short)0, acc[rt][ct], false, false);
    }
    float* cb = ctx + (size_t)(b * NHEADS + head) * DH * DH;
    #pragma unroll
    for (int rt = 0; rt < 2; ++rt)
        #pragma unroll
        for (int ct = 0; ct < 2; ++ct)
            #pragma unroll
            for (int r = 0; r < 8; ++r) {
                const int d = rt * 16 + r + 8 * hi;
                const int e = ct * 16 + lo;
                atomicAdd(&cb[d * DH + e], acc[rt][ct][r]);
            }
}

// ============ Kernel 5: attn[e,n] = sum_d ctx[d,e]*q[d,n]  (single K=32 step) ============
__global__ __launch_bounds__(256)
void k_attn_out(const half_t* __restrict__ qkv_h, const float* __restrict__ ctx,
                half_t* __restrict__ attn_h)
{
    const int wave = blockIdx.x * 8 + (threadIdx.x >> 5);
    const int lane = threadIdx.x & 31;
    const int lo = lane & 15, hi = lane >> 4;
    const int cg = wave & 63;
    const int et = (wave >> 6) & 1;
    const int head = (wave >> 7) & 3;
    const int b = wave >> 9;
    const float* cb = ctx + (size_t)(b * NHEADS + head) * DH * DH;
    const size_t qbase = (size_t)(b * OC3 + head * DH) * NPIX;

    v16h a;
    #pragma unroll
    for (int j = 0; j < 16; ++j)
        a[j] = (half_t)cb[ka_idx(hi, j) * DH + (et * 16 + lo)];

    half_t* ob = attn_h + (size_t)(b * HID + head * DH + et * 16) * NPIX;
    #pragma unroll
    for (int t = 0; t < 4; ++t) {
        const int col = cg * 64 + t * 16 + lo;
        v16h bf;
        #pragma unroll
        for (int j = 0; j < 16; ++j)
            bf[j] = qkv_h[qbase + (size_t)kb_idx(hi, j) * NPIX + col];
        v8f acc = {};
        acc = __builtin_amdgcn_wmma_f32_16x16x32_f16(false, a, false, bf,
                                                     (short)0, acc, false, false);
        #pragma unroll
        for (int r = 0; r < 8; ++r)
            ob[(size_t)(r + 8 * hi) * NPIX + col] = (half_t)acc[r];
    }
}

// ============ Kernel 6: z = W_out[256,128] @ attn + b_out -> d_out (f32) ============
// Same double-buffered TDM structure; B operand already f16 (data_size=2B).
__global__ __launch_bounds__(256)
void k_out_conv(const half_t* __restrict__ attn_h, const float* __restrict__ Wout,
                const float* __restrict__ bout, float* __restrict__ z)
{
    __shared__ half_t xrawh[2][32 * 64]; // 8 KB [k][n], double-buffered TDM dest
    __shared__ half_t lb[64 * 32];       // 4 KB [n][k]
    __shared__ half_t la[128 * 32];      // 8 KB [row][k]

    const int cg = blockIdx.x & 63;
    const int rg = (blockIdx.x >> 6) & 1;           // 128-row group of 256
    const int b  = blockIdx.x >> 7;
    const int tid  = threadIdx.x;
    const int wid  = tid >> 5;
    const int lane = tid & 31;
    const int lo = lane & 15, hi = lane >> 4;
    const int c0 = cg * 64;
    const int rbase = rg * 128;
    const half_t* ab = attn_h + (size_t)b * HID * NPIX;

#if HAVE_TDM
    if (tid < 32)
        tdm_load_2d((unsigned)(uintptr_t)xrawh[0], ab + c0,
                    64u, 32u, (unsigned)NPIX, (unsigned)NPIX, (unsigned)NPIX, 1u);
#endif

    v8f acc[4] = {};
    for (int ks = 0; ks < 4; ++ks) {
        const int k0 = ks * 32;
        const int cur = ks & 1;
#if HAVE_TDM
        if (tid < 32) {
            if (ks < 3) {
                tdm_load_2d((unsigned)(uintptr_t)xrawh[cur ^ 1],
                            ab + (size_t)(k0 + 32) * NPIX + c0,
                            64u, 32u, (unsigned)NPIX, (unsigned)NPIX, (unsigned)NPIX, 1u);
                __builtin_amdgcn_s_wait_tensorcnt(1);
            } else {
                __builtin_amdgcn_s_wait_tensorcnt(0);
            }
        }
#else
        #pragma unroll
        for (int i = 0; i < 8; ++i) {
            const int idx = i * 256 + tid;
            xrawh[cur][idx] = ab[(size_t)(k0 + (idx >> 6)) * NPIX + c0 + (idx & 63)];
        }
#endif
        // W_out tile -> LDS f16: float4 loads, packed v4h stores (128 rows x 32 k)
        #pragma unroll
        for (int i = 0; i < 4; ++i) {
            const int idx4 = i * 256 + tid;
            const int row = idx4 >> 3, koff = (idx4 & 7) * 4;
            const float4 w = *(const float4*)(Wout + (size_t)(rbase + row) * HID + k0 + koff);
            v4h p = {(half_t)w.x, (half_t)w.y, (half_t)w.z, (half_t)w.w};
            *(v4h*)(la + row * 32 + koff) = p;
        }
        __syncthreads();
        // transpose f16: [k][n] -> [n][k], packed b32 stores
        const half_t* xr = xrawh[cur];
        #pragma unroll
        for (int i = 0; i < 4; ++i) {
            const int idx = i * 256 + tid;
            const int n = idx & 63, k = (idx >> 6) * 2;
            v2h p = {xr[k * 64 + n], xr[(k + 1) * 64 + n]};
            *(v2h*)(lb + n * 32 + k) = p;
        }
        __syncthreads();
        const v16h a = frag_a_lds(la, wid * 16 + lo, hi);
        #pragma unroll
        for (int t = 0; t < 4; ++t) {
            const v16h bf = frag_b_lds(lb, t * 16 + lo, hi);
            acc[t] = __builtin_amdgcn_wmma_f32_16x16x32_f16(false, a, false, bf,
                                                            (short)0, acc[t], false, false);
        }
        __syncthreads();
    }
    float* zb = z + (size_t)b * DIMC * NPIX;
    #pragma unroll
    for (int t = 0; t < 4; ++t) {
        const int col = c0 + t * 16 + lo;
        #pragma unroll
        for (int r = 0; r < 8; ++r) {
            const int orow = rbase + wid * 16 + r + 8 * hi;
            zb[(size_t)orow * NPIX + col] = acc[t][r] + bout[orow];
        }
    }
}

// ============ Kernel 7: GroupNorm stats (sum, sumsq) per batch ============
__global__ __launch_bounds__(256)
void k_gn_stats(const float* __restrict__ z, float* __restrict__ stats)
{
    __shared__ float s1[256], s2[256];
    const int b = blockIdx.x >> 6;
    const int blk = blockIdx.x & 63;
    const int tid = threadIdx.x;
    const float* zb = z + (size_t)b * DIMC * NPIX;
    float a1 = 0.f, a2 = 0.f;
    const int base = blk * 256 + tid;
    #pragma unroll 4
    for (int j = 0; j < 64; ++j) {
        const float v = zb[base + j * 16384];
        a1 += v; a2 += v * v;
    }
    s1[tid] = a1; s2[tid] = a2; __syncthreads();
    for (int s = 128; s > 0; s >>= 1) {
        if (tid < s) { s1[tid] += s1[tid + s]; s2[tid] += s2[tid + s]; }
        __syncthreads();
    }
    if (tid == 0) { atomicAdd(&stats[b * 2], s1[0]); atomicAdd(&stats[b * 2 + 1], s2[0]); }
}

// ============ Kernel 8: normalize + per-channel affine ============
__global__ __launch_bounds__(256)
void k_gn_norm(float* __restrict__ z, const float* __restrict__ stats,
               const float* __restrict__ gamma, const float* __restrict__ beta)
{
    const size_t idx = (size_t)blockIdx.x * 256 + threadIdx.x;
    const int b = (int)(idx >> 20);
    const int c = (int)((idx >> 12) & 255);
    const float cnt = (float)(DIMC * NPIX);
    const float mean = stats[b * 2] / cnt;
    const float var  = stats[b * 2 + 1] / cnt - mean * mean;
    const float inv  = rsqrtf(var + GN_EPS);
    z[idx] = (z[idx] - mean) * inv * gamma[c] + beta[c];
}

extern "C" void kernel_launch(void* const* d_in, const int* in_sizes, int n_in,
                              void* d_out, int out_size, void* d_ws, size_t ws_size,
                              hipStream_t stream)
{
    const float* x     = (const float*)d_in[0];
    const float* Wqkv  = (const float*)d_in[1];
    const float* Wout  = (const float*)d_in[2];
    const float* bout  = (const float*)d_in[3];
    const float* gamma = (const float*)d_in[4];
    const float* beta  = (const float*)d_in[5];
    float* out = (float*)d_out;

    char* ws = (char*)d_ws;
    half_t* qkv_h  = (half_t*)(ws);                                   // 50,331,648 B
    float*  ctx    = (float*)(ws + 50331648);                         //    262,144 B
    half_t* attn_h = (half_t*)(ws + 50331648 + 262144);               // 16,777,216 B
    float*  stats  = (float*)(ws + 50331648 + 262144 + 16777216);     //        128 B

    (void)hipMemsetAsync(ctx, 0, 262144, stream);
    (void)hipMemsetAsync(stats, 0, 128, stream);

    k_qkv_gemm <<<3072,  256, 0, stream>>>(x, Wqkv, qkv_h);
    k_q_softmax<<<1024,  256, 0, stream>>>(qkv_h);
    k_k_softmax<<<2048,  256, 0, stream>>>(qkv_h);
    k_context  <<<128,   256, 0, stream>>>(qkv_h, ctx);
    k_attn_out <<<1024,  256, 0, stream>>>(qkv_h, ctx, attn_h);
    k_out_conv <<<2048,  256, 0, stream>>>(attn_h, Wout, bout, out);
    k_gn_stats <<<1024,  256, 0, stream>>>(out, stats);
    k_gn_norm  <<<65536, 256, 0, stream>>>(out, stats, gamma, beta);
}
